// TimeSeriesModel_16681652978332
// MI455X (gfx1250) — compile-verified, matrix operand order checked by reading, not run on hardware
//
#include <hip/hip_runtime.h>
#include <hip/hip_bf16.h>
#include <math.h>

typedef __bf16 bf16;
typedef __attribute__((ext_vector_type(16))) __bf16 v16bf;
typedef __attribute__((ext_vector_type(8)))  __bf16 v8bf;
typedef __attribute__((ext_vector_type(8)))  float  v8f;

constexpr int NB    = 512;       // batch
constexpr int NT    = 64;        // timesteps
constexpr int NNODE = 13;
constexpr int NBT   = NB * NT;        // 32768
constexpr int NBTN  = NBT * NNODE;    // 425984
constexpr int HIDN  = 256;
constexpr int G3    = 768;            // 3*HID gates

// edge list: PHYSIO both directions + self loops (E=45)
__constant__ int d_src[45] = {0,7,0,10,0,6,0,4,0,8,0,11,1,7,2,9,3,4,4,10,5,6,6,10,8,9,8,10,10,11,11,12,
                              0,1,2,3,4,5,6,7,8,9,10,11,12};
__constant__ int d_dst[45] = {7,0,10,0,6,0,4,0,8,0,11,0,7,1,9,2,4,3,10,4,6,5,10,6,9,8,10,8,11,10,12,11,
                              0,1,2,3,4,5,6,7,8,9,10,11,12};

// ---------------------------------------------------------------------------
// Packed-B fragment layout: for tileN (16 cols), K-chunk kc (32 deep), lane:
//   o[(((tileN*(K/32))+kc)*32 + lane)*16 + j]  holds the j-th bf16 of that
// lane's B fragment for v_wmma_f32_16x16x32_bf16. One v16bf load per K-step.
// ---------------------------------------------------------------------------
__device__ __forceinline__ void frag_decode(int i, int K, int& k, int& n)
{
  const int j = i & 15, lane = (i >> 4) & 31, chunk = i >> 9;
  const int KC = K >> 5;
  const int tileN = chunk / KC, kc = chunk % KC;
  n = tileN * 16 + (lane & 15);
  const int kb = (lane >> 4) * 8;
  const int v = j >> 1, odd = j & 1;
  const int kk = (v < 4) ? (kb + 2 * v + odd) : (16 + kb + 2 * (v - 4) + odd);
  k = kc * 32 + kk;
}

// pack row-major f32 W[K,N] into fragment order (bf16)
__global__ void k_packw(const float* __restrict__ in, bf16* __restrict__ o, int K, int N)
{
  const int i = blockIdx.x * blockDim.x + threadIdx.x;
  if (i >= K * N) return;
  int k, n; frag_decode(i, K, k, n);
  o[i] = (bf16)in[k * N + n];
}

// pack torch-style f32 W[N,K] (out,in) into fragment order (bf16) == transpose
__global__ void k_packwT(const float* __restrict__ in, bf16* __restrict__ o, int K, int N)
{
  const int i = blockIdx.x * blockDim.x + threadIdx.x;
  if (i >= K * N) return;
  int k, n; frag_decode(i, K, k, n);
  o[i] = (bf16)in[n * K + k];
}

// fused GAT0 weight: [64,256] = [gat0.w (33x128, zero-pad K) | proj.w], packed
__global__ void k_build_w0(const float* __restrict__ w0, const float* __restrict__ wp,
                           const float* __restrict__ pb,
                           bf16* __restrict__ o, float* __restrict__ bias0)
{
  const int i = blockIdx.x * blockDim.x + threadIdx.x;
  if (i < 64 * 256) {
    int k, n; frag_decode(i, 64, k, n);
    float v = 0.f;
    if (k < 33) v = (n < 128) ? w0[k * 128 + n] : wp[k * 128 + (n - 128)];
    o[i] = (bf16)v;
  }
  if (i < 256) bias0[i] = (i < 128) ? 0.f : pb[i - 128];
}

// ---------------------------------------------------------------------------
// Generic bf16 WMMA GEMM:  C[M,N] = A[M,K] * Wpacked (+bias[N]), f32 accum.
// block = (32,4): 4 waves, each one 16x16 tile. grid = (M/64, N/16).
// Requires M%64==0, K%32==0, N%16==0. B loads: 1 v16bf per K-step.
// ---------------------------------------------------------------------------
__global__ __launch_bounds__(128)
void k_gemm_bf16(const bf16* __restrict__ A, const bf16* __restrict__ Wp,
                 const float* __restrict__ bias, float* __restrict__ C,
                 int M, int N, int K)
{
  const int lane  = threadIdx.x;                    // 0..31
  const int tileM = blockIdx.x * 4 + threadIdx.y;
  const int tileN = blockIdx.y;
  const int row0  = tileM * 16, col0 = tileN * 16;
  const int mr = lane & 15;
  const int kb = (lane >> 4) * 8;
  const int hi = lane >> 4;
  v8f acc = {};
  const bf16* Arow  = A + (size_t)(row0 + mr) * K;
  const bf16* Bfrag = Wp + ((size_t)tileN * (K >> 5) * 32 + lane) * 16;
  for (int k0 = 0; k0 < K; k0 += 32) {
    const v8bf alo = *(const v8bf*)(Arow + k0 + kb);
    const v8bf ahi = *(const v8bf*)(Arow + k0 + 16 + kb);
    const v16bf fa = __builtin_shufflevector(alo, ahi,
        0, 1, 2, 3, 4, 5, 6, 7, 8, 9, 10, 11, 12, 13, 14, 15);
    const v16bf fb = *(const v16bf*)Bfrag;
    Bfrag += 32 * 16;
    acc = __builtin_amdgcn_wmma_f32_16x16x32_bf16(false, fa, false, fb,
                                                  (short)0, acc, false, false);
  }
  const int nc = col0 + mr;
  const float bv = bias ? bias[nc] : 0.f;
#pragma unroll
  for (int r = 0; r < 8; ++r)
    C[(size_t)(row0 + r + 8 * hi) * N + nc] = acc[r] + bv;
}

// ---------------------------------------------------------------------------
// Fused GRU recurrent step: hg = h_prev @ whh^T + bhh (WMMA into LDS), then
// r/z/n gates + state update. One block owns 16 batch rows (no cross-block
// dependency); 8 waves x 6 col-tiles cover N=768. whh pre-packed (K=256,N=768).
// ---------------------------------------------------------------------------
__global__ __launch_bounds__(256)
void k_gru_step(const bf16* __restrict__ hprev_b, const float* __restrict__ hprev_f,
                const bf16* __restrict__ whhP,    // packed [256,768]
                const float* __restrict__ bhh,    // [768]
                const float* __restrict__ xg,     // [B,T,768]
                int t,
                float* __restrict__ outseq,       // [B,T,512]
                int outoff,
                float* __restrict__ hnew_f, bf16* __restrict__ hnew_b)
{
  __shared__ float hg[16 * G3];                   // 48 KB
  const int lane = threadIdx.x, wv = threadIdx.y;
  const int row0 = blockIdx.x * 16;
  const int mr = lane & 15, kb = (lane >> 4) * 8, hi = lane >> 4;
  const bf16* Arow = hprev_b + (size_t)(row0 + mr) * HIDN;
  for (int nt = 0; nt < 6; ++nt) {
    const int tileN = wv * 6 + nt;
    const int col0  = tileN * 16;
    const bf16* Bfrag = whhP + ((size_t)tileN * 8 * 32 + lane) * 16;  // K/32 = 8
    v8f acc = {};
    for (int k0 = 0; k0 < HIDN; k0 += 32) {
      const v8bf alo = *(const v8bf*)(Arow + k0 + kb);
      const v8bf ahi = *(const v8bf*)(Arow + k0 + 16 + kb);
      const v16bf fa = __builtin_shufflevector(alo, ahi,
          0, 1, 2, 3, 4, 5, 6, 7, 8, 9, 10, 11, 12, 13, 14, 15);
      const v16bf fb = *(const v16bf*)Bfrag;
      Bfrag += 32 * 16;
      acc = __builtin_amdgcn_wmma_f32_16x16x32_bf16(false, fa, false, fb,
                                                    (short)0, acc, false, false);
    }
#pragma unroll
    for (int r = 0; r < 8; ++r)
      hg[(r + 8 * hi) * G3 + col0 + mr] = acc[r] + bhh[col0 + mr];
  }
  __syncthreads();
  const int tid = wv * 32 + lane;
  for (int i = tid; i < 16 * HIDN; i += 256) {
    const int r = i >> 8, u = i & 255;
    const int b = row0 + r;
    const float* xr = xg + ((size_t)b * NT + t) * G3;
    const float hr = hg[r * G3 + u], hz = hg[r * G3 + 256 + u], hn = hg[r * G3 + 512 + u];
    const float rr = 1.f / (1.f + __expf(-(xr[u]       + hr)));
    const float zz = 1.f / (1.f + __expf(-(xr[256 + u] + hz)));
    const float nn = tanhf(xr[512 + u] + rr * hn);
    const float hp = hprev_f[(size_t)b * HIDN + u];
    const float hv = (1.f - zz) * nn + zz * hp;
    hnew_f[(size_t)b * HIDN + u] = hv;
    hnew_b[(size_t)b * HIDN + u] = (bf16)hv;
    outseq[((size_t)b * NT + t) * 512 + outoff + u] = hv;
  }
}

// ---------------------------------------------------------------------------
// GAT attention per (b,t): scatter-softmax over 45 edges, aggregate, +bias.
// ---------------------------------------------------------------------------
__global__ __launch_bounds__(64)
void k_gat_attn(const float* __restrict__ Hbuf, int ldH,
                const float* __restrict__ asrc, const float* __restrict__ adst,
                const float* __restrict__ bias, float* __restrict__ conv)
{
  __shared__ float sh[13 * 128];
  __shared__ float sas[13][4], sad[13][4];
  __shared__ float salpha[45][4];
  const size_t base = (size_t)blockIdx.x * NNODE;
  const int tid = threadIdx.x;
  for (int i = tid; i < 13 * 128; i += 64)
    sh[i] = Hbuf[(base + (i >> 7)) * (size_t)ldH + (i & 127)];
  __syncthreads();
  if (tid < 52) {
    const int n = tid >> 2, h = tid & 3;
    float s = 0.f, d = 0.f;
    for (int c = 0; c < 32; ++c) {
      const float hv = sh[n * 128 + h * 32 + c];
      s += hv * asrc[h * 32 + c];
      d += hv * adst[h * 32 + c];
    }
    sas[n][h] = s; sad[n][h] = d;
  }
  __syncthreads();
  if (tid < 52) {
    const int n = tid >> 2, h = tid & 3;
    float mx = -1e30f;
    for (int e = 0; e < 45; ++e) if (d_dst[e] == n) {
      float v = sas[d_src[e]][h] + sad[n][h];
      v = v >= 0.f ? v : 0.2f * v;
      mx = fmaxf(mx, v);
    }
    float den = 0.f;
    for (int e = 0; e < 45; ++e) if (d_dst[e] == n) {
      float v = sas[d_src[e]][h] + sad[n][h];
      v = v >= 0.f ? v : 0.2f * v;
      const float ex = __expf(v - mx);
      salpha[e][h] = ex; den += ex;
    }
    const float inv = 1.f / den;
    for (int e = 0; e < 45; ++e) if (d_dst[e] == n) salpha[e][h] *= inv;
  }
  __syncthreads();
  for (int i = tid; i < 13 * 128; i += 64) {
    const int dn = i >> 7, hc = i & 127, h = hc >> 5;
    float acc = bias[hc];
    for (int e = 0; e < 45; ++e) if (d_dst[e] == dn)
      acc += salpha[e][h] * sh[d_src[e] * 128 + hc];
    conv[(base + dn) * 128 + hc] = acc;
  }
}

// relu(LN(x) + res); optional bf16 mirror output. one block per 128-wide row.
__global__ __launch_bounds__(128)
void k_ln_add_relu(const float* __restrict__ X, const float* __restrict__ res, int ldres,
                   const float* __restrict__ g, const float* __restrict__ b,
                   float* __restrict__ outf, bf16* __restrict__ outb)
{
  __shared__ float red[128];
  const size_t row = blockIdx.x;
  const int c = threadIdx.x;
  const float x = X[row * 128 + c];
  red[c] = x; __syncthreads();
  for (int s = 64; s > 0; s >>= 1) { if (c < s) red[c] += red[c + s]; __syncthreads(); }
  const float mu = red[0] * (1.f / 128.f); __syncthreads();
  const float dv = x - mu;
  red[c] = dv * dv; __syncthreads();
  for (int s = 64; s > 0; s >>= 1) { if (c < s) red[c] += red[c + s]; __syncthreads(); }
  const float var = red[0] * (1.f / 128.f);
  float y = g[c] * dv * rsqrtf(var + 1e-5f) + b[c] + res[row * (size_t)ldres + c];
  y = fmaxf(y, 0.f);
  outf[row * 128 + c] = y;
  if (outb) outb[row * 128 + c] = (bf16)y;
}

// node features: [x | 32-dim sin/cos pos enc | zero-pad to 64], bf16
__global__ void k_pos_node(const float* __restrict__ x, const int* __restrict__ ts,
                           bf16* __restrict__ nodeb)
{
  const int i = blockIdx.x * blockDim.x + threadIdx.x;
  if (i >= NBTN) return;
  const int n = i % NNODE, bt = i / NNODE;
  const float tv = (float)ts[bt];
  bf16* o = nodeb + (size_t)i * 64;
  o[0] = (bf16)x[bt * NNODE + n];
#pragma unroll
  for (int p = 0; p < 16; ++p) {
    const float div = __expf(-(float)(2 * p) * 0.28782313662425572f); // ln(1e4)/32
    const float ang = tv * div;
    o[1 + 2 * p] = (bf16)sinf(ang);
    o[2 + 2 * p] = (bf16)cosf(ang);
  }
  for (int q = 33; q < 64; ++q) o[q] = (bf16)0.f;
}

// raw = gelu(BN(x @ W13x128 + b))
__global__ void k_raw(const float* __restrict__ x, const float* __restrict__ w,
                      const float* __restrict__ b,
                      const float* __restrict__ bg, const float* __restrict__ bb,
                      const float* __restrict__ bm, const float* __restrict__ bv,
                      float* __restrict__ raw)
{
  const int i = blockIdx.x * blockDim.x + threadIdx.x;
  if (i >= NBT * 128) return;
  const int bt = i >> 7, o = i & 127;
  const float* xr = x + bt * NNODE;
  float acc = b[o];
  for (int j = 0; j < NNODE; ++j) acc += xr[j] * w[j * 128 + o];
  acc = bg[o] * (acc - bm[o]) * rsqrtf(bv[o] + 1e-5f) + bb[o];
  raw[i] = 0.5f * acc * (1.f + erff(acc * 0.70710678118654752f));
}

// sfeat = lrelu(BN(static @ W7x64 + b), 0.01)
__global__ void k_static(const float* __restrict__ st, const float* __restrict__ w,
                         const float* __restrict__ b,
                         const float* __restrict__ bg, const float* __restrict__ bb,
                         const float* __restrict__ bm, const float* __restrict__ bv,
                         float* __restrict__ sf)
{
  const int i = blockIdx.x * blockDim.x + threadIdx.x;
  if (i >= NB * 64) return;
  const int bq = i >> 6, o = i & 63;
  float acc = b[o];
  for (int j = 0; j < 7; ++j) acc += st[bq * 7 + j] * w[j * 64 + o];
  acc = bg[o] * (acc - bm[o]) * rsqrtf(bv[o] + 1e-5f) + bb[o];
  sf[i] = acc >= 0.f ? acc : 0.01f * acc;
}

__global__ void k_gatseq(const float* __restrict__ g2, float* __restrict__ gs)
{
  const int i = blockIdx.x * blockDim.x + threadIdx.x;
  if (i >= NBT * 128) return;
  const int bt = i >> 7, c = i & 127;
  float s = 0.f;
  for (int n = 0; n < NNODE; ++n) s += g2[((size_t)bt * NNODE + n) * 128 + c];
  gs[i] = s * (1.f / 13.f);
}

__global__ void k_ffb(const float* __restrict__ raw, const float* __restrict__ gs,
                      bf16* __restrict__ ffb)
{
  const int i = blockIdx.x * blockDim.x + threadIdx.x;
  if (i >= NBT * 256) return;
  const int bt = i >> 8, c = i & 255;
  ffb[i] = (bf16)(c < 128 ? raw[bt * 128 + c] : gs[bt * 128 + c - 128]);
}

__global__ void k_fuse(const float* __restrict__ gateb, const float* __restrict__ raw,
                       const float* __restrict__ gs, const float* __restrict__ sf,
                       bf16* __restrict__ fb)
{
  const int i = blockIdx.x * blockDim.x + threadIdx.x;
  if (i >= NBT * 320) return;
  const int bt = i / 320, c = i % 320;
  float v;
  if (c < 256) {
    const float g = 1.f / (1.f + __expf(-gateb[(size_t)bt * 256 + c]));
    v = g * (c < 128 ? raw[bt * 128 + c] : gs[bt * 128 + (c - 128)]);
  } else {
    v = sf[(bt / NT) * 64 + (c - 256)];
  }
  fb[i] = (bf16)v;
}

__global__ void k_f2b(const float* __restrict__ in, bf16* __restrict__ o, int n)
{
  const int i = blockIdx.x * blockDim.x + threadIdx.x;
  if (i < n) o[i] = (bf16)in[i];
}

__global__ void k_zero_h(float* __restrict__ hf, bf16* __restrict__ hb, int n)
{
  const int i = blockIdx.x * blockDim.x + threadIdx.x;
  if (i < n) { hf[i] = 0.f; hb[i] = (bf16)0.f; }
}

// head: LN(last step) -> residual block -> BN -> lrelu -> out linear
__global__ __launch_bounds__(128)
void k_head(const float* __restrict__ out1,
            const float* __restrict__ lng, const float* __restrict__ lnb,
            const float* __restrict__ rbg, const float* __restrict__ rbb,
            const float* __restrict__ rbm, const float* __restrict__ rbv,
            const float* __restrict__ rw,  const float* __restrict__ rb,
            const float* __restrict__ sw,  const float* __restrict__ sb,
            const float* __restrict__ obg, const float* __restrict__ obb,
            const float* __restrict__ obm, const float* __restrict__ obv,
            const float* __restrict__ ow,  const float* __restrict__ ob,
            float* __restrict__ out)
{
  __shared__ float sf[512], sa[512], sr[128], red[128];
  const int b = blockIdx.x, tid = threadIdx.x;
  const float* f = out1 + ((size_t)b * NT + (NT - 1)) * 512;
  float loc[4]; float s = 0.f;
#pragma unroll
  for (int j = 0; j < 4; ++j) { loc[j] = f[tid * 4 + j]; s += loc[j]; }
  red[tid] = s; __syncthreads();
  for (int st = 64; st > 0; st >>= 1) { if (tid < st) red[tid] += red[tid + st]; __syncthreads(); }
  const float mu = red[0] * (1.f / 512.f); __syncthreads();
  s = 0.f;
#pragma unroll
  for (int j = 0; j < 4; ++j) { const float d = loc[j] - mu; s += d * d; }
  red[tid] = s; __syncthreads();
  for (int st = 64; st > 0; st >>= 1) { if (tid < st) red[tid] += red[tid + st]; __syncthreads(); }
  const float rstd = rsqrtf(red[0] * (1.f / 512.f) + 1e-5f);
#pragma unroll
  for (int j = 0; j < 4; ++j) {
    const int c = tid * 4 + j;
    const float fn = lng[c] * (loc[j] - mu) * rstd + lnb[c];
    sf[c] = fn;
    const float a = rbg[c] * (fn - rbm[c]) * rsqrtf(rbv[c] + 1e-5f) + rbb[c];
    sa[c] = fmaxf(a, 0.f);
  }
  __syncthreads();
  float acc = rb[tid] + sb[tid];
  for (int c = 0; c < 512; ++c) acc += sa[c] * rw[c * 128 + tid] + sf[c] * sw[c * 128 + tid];
  const float r2 = obg[tid] * (acc - obm[tid]) * rsqrtf(obv[tid] + 1e-5f) + obb[tid];
  sr[tid] = r2 >= 0.f ? r2 : 0.01f * r2;
  __syncthreads();
  if (tid < 13) {
    float o = ob[tid];
    for (int c = 0; c < 128; ++c) o += sr[c] * ow[c * 13 + tid];
    out[b * 13 + tid] = o;
  }
}

// ---------------------------------------------------------------------------
// input indices (setup_inputs dict insertion order, params leaves nested)
// ---------------------------------------------------------------------------
enum {
  IN_X = 0, IN_STATIC,
  P_GAT0_W, P_GAT0_ASRC, P_GAT0_ADST, P_GAT0_BIAS,
  P_GAT0LN_G, P_GAT0LN_B,
  P_PROJ_W, P_PROJ_B,
  P_GAT1_W, P_GAT1_ASRC, P_GAT1_ADST, P_GAT1_BIAS,
  P_GAT1LN_G, P_GAT1LN_B,
  P_RAW_W, P_RAW_B,
  P_RAWBN_G, P_RAWBN_B, P_RAWBN_M, P_RAWBN_V,
  P_ST_W, P_ST_B,
  P_STBN_G, P_STBN_B, P_STBN_M, P_STBN_V,
  P_GATE_W, P_GATE_B,
  P_G0F_WIH, P_G0F_WHH, P_G0F_BIH, P_G0F_BHH,
  P_G0B_WIH, P_G0B_WHH, P_G0B_BIH, P_G0B_BHH,
  P_G1F_WIH, P_G1F_WHH, P_G1F_BIH, P_G1F_BHH,
  P_G1B_WIH, P_G1B_WHH, P_G1B_BIH, P_G1B_BHH,
  P_GRULN_G, P_GRULN_B,
  P_RESBN_G, P_RESBN_B, P_RESBN_M, P_RESBN_V,
  P_RES_W, P_RES_B,
  P_SHORT_W, P_SHORT_B,
  P_OUTBN_G, P_OUTBN_B, P_OUTBN_M, P_OUTBN_V,
  P_OUT_W, P_OUT_B,
  IN_TS
};

extern "C" void kernel_launch(void* const* d_in, const int* in_sizes, int n_in,
                              void* d_out, int out_size, void* d_ws, size_t ws_size,
                              hipStream_t stream)
{
  (void)in_sizes; (void)n_in; (void)out_size; (void)ws_size;
  auto F = [&](int i) { return (const float*)d_in[i]; };
  const int* ts = (const int*)d_in[IN_TS];
  float* outp = (float*)d_out;

  char* wsp = (char*)d_ws;
  auto alloc = [&](size_t bytes) -> char* {
    char* p = wsp; wsp += (bytes + 255) & ~(size_t)255; return p;
  };

  bf16*  nodeb  = (bf16*)alloc((size_t)NBTN * 64 * 2);
  bf16*  wcat0  = (bf16*)alloc(64 * 256 * 2);
  float* bias0  = (float*)alloc(256 * 4);
  bf16*  w1b    = (bf16*)alloc(128 * 128 * 2);
  bf16*  gatewb = (bf16*)alloc(256 * 256 * 2);
  bf16*  wih0fT = (bf16*)alloc(320 * 768 * 2);
  bf16*  wih0bT = (bf16*)alloc(320 * 768 * 2);
  bf16*  wih1fT = (bf16*)alloc(512 * 768 * 2);
  bf16*  wih1bT = (bf16*)alloc(512 * 768 * 2);
  bf16*  whh0fT = (bf16*)alloc(256 * 768 * 2);
  bf16*  whh0bT = (bf16*)alloc(256 * 768 * 2);
  bf16*  whh1fT = (bf16*)alloc(256 * 768 * 2);
  bf16*  whh1bT = (bf16*)alloc(256 * 768 * 2);
  float* hp     = (float*)alloc((size_t)NBTN * 256 * 4);   // [h0|proj], later h1 + g2
  float* conv   = (float*)alloc((size_t)NBTN * 128 * 4);
  float* gbuf   = (float*)alloc((size_t)NBTN * 128 * 4);
  bf16*  gbf    = (bf16*)alloc((size_t)NBTN * 128 * 2);
  float* g2     = hp + (size_t)NBTN * 128;                 // reuse hp's 2nd half
  float* rawb   = (float*)alloc((size_t)NBT * 128 * 4);
  float* gseq   = (float*)alloc((size_t)NBT * 128 * 4);
  bf16*  ffb    = (bf16*)alloc((size_t)NBT * 256 * 2);
  float* gateb  = (float*)alloc((size_t)NBT * 256 * 4);
  bf16*  fusedb = (bf16*)alloc((size_t)NBT * 320 * 2);
  float* sfeat  = (float*)alloc(NB * 64 * 4);
  float* xgf    = (float*)alloc((size_t)NBT * 768 * 4);
  float* xgb    = (float*)alloc((size_t)NBT * 768 * 4);
  float* out0   = (float*)alloc((size_t)NBT * 512 * 4);
  bf16*  out0b  = (bf16*)alloc((size_t)NBT * 512 * 2);
  float* out1   = (float*)alloc((size_t)NBT * 512 * 4);
  float* hf0    = (float*)alloc(NB * HIDN * 4);
  float* hf1    = (float*)alloc(NB * HIDN * 4);
  bf16*  hb0    = (bf16*)alloc(NB * HIDN * 2);
  bf16*  hb1    = (bf16*)alloc(NB * HIDN * 2);

  auto gemm = [&](const bf16* A, const bf16* W, const float* bias, float* C,
                  int M, int N, int K) {
    k_gemm_bf16<<<dim3(M / 64, N / 16), dim3(32, 4), 0, stream>>>(A, W, bias, C, M, N, K);
  };

  // ---- weight prep (f32 -> packed bf16 fragment layouts) ----
  k_build_w0<<<(64 * 256 + 255) / 256, 256, 0, stream>>>(F(P_GAT0_W), F(P_PROJ_W), F(P_PROJ_B), wcat0, bias0);
  k_packw<<<(128 * 128 + 255) / 256, 256, 0, stream>>>(F(P_GAT1_W), w1b, 128, 128);
  k_packw<<<(256 * 256 + 255) / 256, 256, 0, stream>>>(F(P_GATE_W), gatewb, 256, 256);
  k_packwT<<<(768 * 320 + 255) / 256, 256, 0, stream>>>(F(P_G0F_WIH), wih0fT, 320, 768);
  k_packwT<<<(768 * 320 + 255) / 256, 256, 0, stream>>>(F(P_G0B_WIH), wih0bT, 320, 768);
  k_packwT<<<(768 * 512 + 255) / 256, 256, 0, stream>>>(F(P_G1F_WIH), wih1fT, 512, 768);
  k_packwT<<<(768 * 512 + 255) / 256, 256, 0, stream>>>(F(P_G1B_WIH), wih1bT, 512, 768);
  k_packwT<<<(768 * 256 + 255) / 256, 256, 0, stream>>>(F(P_G0F_WHH), whh0fT, 256, 768);
  k_packwT<<<(768 * 256 + 255) / 256, 256, 0, stream>>>(F(P_G0B_WHH), whh0bT, 256, 768);
  k_packwT<<<(768 * 256 + 255) / 256, 256, 0, stream>>>(F(P_G1F_WHH), whh1fT, 256, 768);
  k_packwT<<<(768 * 256 + 255) / 256, 256, 0, stream>>>(F(P_G1B_WHH), whh1bT, 256, 768);

  // ---- node features + GAT block 0 ----
  k_pos_node<<<(NBTN + 255) / 256, 256, 0, stream>>>(F(IN_X), ts, nodeb);
  gemm(nodeb, wcat0, bias0, hp, NBTN, 256, 64);                       // [h0 | proj]
  k_gat_attn<<<NBT, 64, 0, stream>>>(hp, 256, F(P_GAT0_ASRC), F(P_GAT0_ADST), F(P_GAT0_BIAS), conv);
  k_ln_add_relu<<<NBTN, 128, 0, stream>>>(conv, hp + 128, 256, F(P_GAT0LN_G), F(P_GAT0LN_B), gbuf, gbf);

  // ---- GAT block 1 ----
  gemm(gbf, w1b, nullptr, hp, NBTN, 128, 128);                        // h1 (overwrites hp)
  k_gat_attn<<<NBT, 64, 0, stream>>>(hp, 128, F(P_GAT1_ASRC), F(P_GAT1_ADST), F(P_GAT1_BIAS), conv);
  k_ln_add_relu<<<NBTN, 128, 0, stream>>>(conv, gbuf, 128, F(P_GAT1LN_G), F(P_GAT1LN_B), g2, nullptr);

  // ---- sequence features, gate, fuse ----
  k_gatseq<<<(NBT * 128 + 255) / 256, 256, 0, stream>>>(g2, gseq);
  k_raw<<<(NBT * 128 + 255) / 256, 256, 0, stream>>>(F(IN_X), F(P_RAW_W), F(P_RAW_B),
      F(P_RAWBN_G), F(P_RAWBN_B), F(P_RAWBN_M), F(P_RAWBN_V), rawb);
  k_static<<<(NB * 64 + 255) / 256, 256, 0, stream>>>(F(IN_STATIC), F(P_ST_W), F(P_ST_B),
      F(P_STBN_G), F(P_STBN_B), F(P_STBN_M), F(P_STBN_V), sfeat);
  k_ffb<<<(NBT * 256 + 255) / 256, 256, 0, stream>>>(rawb, gseq, ffb);
  gemm(ffb, gatewb, F(P_GATE_B), gateb, NBT, 256, 256);
  k_fuse<<<(NBT * 320 + 255) / 256, 256, 0, stream>>>(gateb, rawb, gseq, sfeat, fusedb);

  // ---- GRU layers ----
  auto run_dir = [&](const float* xg, const bf16* whhP, const float* bhh,
                     float* outseq, int outoff, bool reverse) {
    k_zero_h<<<(NB * HIDN + 255) / 256, 256, 0, stream>>>(hf0, hb0, NB * HIDN);
    float* hf[2] = { hf0, hf1 };
    bf16*  hb[2] = { hb0, hb1 };
    for (int s = 0; s < NT; ++s) {
      const int t = reverse ? (NT - 1 - s) : s;
      const int a = s & 1, b = 1 - a;
      k_gru_step<<<dim3(NB / 16), dim3(32, 8), 0, stream>>>(
          hb[a], hf[a], whhP, bhh, xg, t, outseq, outoff, hf[b], hb[b]);
    }
  };

  gemm(fusedb, wih0fT, F(P_G0F_BIH), xgf, NBT, 768, 320);
  gemm(fusedb, wih0bT, F(P_G0B_BIH), xgb, NBT, 768, 320);
  run_dir(xgf, whh0fT, F(P_G0F_BHH), out0, 0,   false);
  run_dir(xgb, whh0bT, F(P_G0B_BHH), out0, 256, true);

  k_f2b<<<(NBT * 512 + 255) / 256, 256, 0, stream>>>(out0, out0b, NBT * 512);
  gemm(out0b, wih1fT, F(P_G1F_BIH), xgf, NBT, 768, 512);
  gemm(out0b, wih1bT, F(P_G1B_BIH), xgb, NBT, 768, 512);
  run_dir(xgf, whh1fT, F(P_G1F_BHH), out1, 0,   false);
  run_dir(xgb, whh1bT, F(P_G1B_BHH), out1, 256, true);

  // ---- head ----
  k_head<<<NB, 128, 0, stream>>>(out1,
      F(P_GRULN_G), F(P_GRULN_B),
      F(P_RESBN_G), F(P_RESBN_B), F(P_RESBN_M), F(P_RESBN_V),
      F(P_RES_W), F(P_RES_B), F(P_SHORT_W), F(P_SHORT_B),
      F(P_OUTBN_G), F(P_OUTBN_B), F(P_OUTBN_M), F(P_OUTBN_V),
      F(P_OUT_W), F(P_OUT_B), outp);
}
